// GCN_graph_42838003810874
// MI455X (gfx1250) — matrix operand order repaired and checked by reading, not compiled, not run on hardware
//
#include <hip/hip_runtime.h>
#include <hip/hip_bf16.h>

typedef __attribute__((ext_vector_type(16))) _Float16 v16h;
typedef __attribute__((ext_vector_type(8)))  _Float16 v8h;
typedef __attribute__((ext_vector_type(4)))  _Float16 v4h;
typedef __attribute__((ext_vector_type(8)))  float    v8f;

#define NFEAT 128
#define NHID  128
#define NCLS  16

// ---------------------------------------------------------------- degrees ---
__global__ void k_init_deg(float* __restrict__ deg, int n) {
    int i = blockIdx.x * blockDim.x + threadIdx.x;
    if (i < n) deg[i] = 1.0f;                       // self loop contributes 1
}

__global__ void k_deg_accum(float* __restrict__ deg, const int* __restrict__ ei, int E) {
    int e = blockIdx.x * blockDim.x + threadIdx.x;
    if (e < E) atomicAdd(&deg[ei[e]], 1.0f);        // row = destination
}

__global__ void k_deg_rsqrt(float* __restrict__ deg, int n) {
    int i = blockIdx.x * blockDim.x + threadIdx.x;
    if (i < n) deg[i] = rsqrtf(fmaxf(deg[i], 1e-12f));
}

// ------------------------------------------------------------ f32 -> f16 ----
__global__ void k_f32_to_f16(const float* __restrict__ src, _Float16* __restrict__ dst,
                             int total4) {
    int i4 = blockIdx.x * blockDim.x + threadIdx.x;
    if (i4 >= total4) return;
    const float4 v = *(const float4*)(src + (size_t)i4 * 4);
    v4h h; h[0] = (_Float16)v.x; h[1] = (_Float16)v.y; h[2] = (_Float16)v.z; h[3] = (_Float16)v.w;
    *(v4h*)(dst + (size_t)i4 * 4) = h;
}

// ---------------------------------------------- pack W into B-fragment order
// B fragment (16-bit, 32x16): lane n<16 holds col n, K = ks*32 + {0..7,16..23}
//                             lane n>=16 holds col n-16, K = ks*32 + {8..15,24..31}
template <int M>
__global__ void k_pack_w(const float* __restrict__ W, _Float16* __restrict__ Bfrag) {
    constexpr int CT = M / 16;
    int t = blockIdx.x * blockDim.x + threadIdx.x;
    if (t >= CT * 4 * 32) return;
    int lane = t & 31;
    int ks   = (t >> 5) & 3;
    int ct   = t >> 7;
    int col  = ct * 16 + (lane & 15);
    int kbase = ks * 32 + ((lane >> 4) ? 8 : 0);
    _Float16* dst = Bfrag + (size_t)t * 16;
#pragma unroll
    for (int i = 0; i < 16; ++i)
        dst[i] = (_Float16)W[(size_t)(kbase + (i < 8 ? i : i + 8)) * M + col];
}

// --------------------------------------------------------------- WMMA GEMM --
// C[N x M] = A[N x 128] * B[128 x M].
// One wave computes a 16 x (16*CPW) strip: A fragment reused CPW times per k-step.
template <int M, int CPW>
__global__ __launch_bounds__(256)
void k_gemm(const _Float16* __restrict__ A, const _Float16* __restrict__ Bfrag,
            float* __restrict__ C, int nRows, int nGroups) {
    constexpr int CT  = M / 16;
    constexpr int GPR = CT / CPW;                   // col groups per row-tile
    int g = (int)(blockIdx.x * (blockDim.x >> 5) + (threadIdx.x >> 5));
    if (g >= nGroups) return;                       // wave-uniform: EXEC all-1s at WMMA
    int lane   = threadIdx.x & 31;
    int laneLo = lane & 15;
    int hi     = lane >> 4;
    int rowTile = g / GPR;
    int colBase = (g % GPR) * CPW;

    int row  = rowTile * 16 + laneLo;
    int rowA = row < nRows ? row : nRows - 1;       // clamp reads; stores guarded
    const _Float16* arow = A + (size_t)rowA * 128 + hi * 8;

    v8f c[CPW] = {};
#pragma unroll
    for (int ks = 0; ks < 4; ++ks) {
        // A fragment: lanes 0-15 row M=l, K = ks*32 + {0..7,16..23}; lanes 16-31 K+8
        v8h alo = *(const v8h*)(arow + ks * 32);
        v8h ahi = *(const v8h*)(arow + ks * 32 + 16);
        v16h a;
#pragma unroll
        for (int i = 0; i < 8; ++i) { a[i] = alo[i]; a[i + 8] = ahi[i]; }
#pragma unroll
        for (int j = 0; j < CPW; ++j) {
            // B fragment: pre-packed contiguous 16 halves per lane
            v16h b = *(const v16h*)(Bfrag +
                     ((size_t)((colBase + j) * 4 + ks) * 32 + lane) * 16);
            c[j] = __builtin_amdgcn_wmma_f32_16x16x32_f16(false, a, false, b, (short)0,
                                                          c[j], false, false);
        }
    }
    // D layout: VGPR r -> row rowTile*16 + r + 8*hi, col = colTile*16 + laneLo
    int rbase = rowTile * 16 + hi * 8;
    if ((rowTile * 16 + 16) <= nRows) {             // wave-uniform fast path: no guards
#pragma unroll
        for (int j = 0; j < CPW; ++j) {
            int col = (colBase + j) * 16 + laneLo;
#pragma unroll
            for (int r = 0; r < 8; ++r)
                C[(size_t)(rbase + r) * M + col] = c[j][r];
        }
    } else {
#pragma unroll
        for (int j = 0; j < CPW; ++j) {
            int col = (colBase + j) * 16 + laneLo;
#pragma unroll
            for (int r = 0; r < 8; ++r) {
                int orow = rbase + r;
                if (orow < nRows) C[(size_t)orow * M + col] = c[j][r];
            }
        }
    }
}

// ------------------------------------------------------------------- zero ---
__global__ void k_zero4(float* __restrict__ p, int total4) {
    int i = blockIdx.x * blockDim.x + threadIdx.x;
    if (i < total4) *(float4*)(p + (size_t)i * 4) = make_float4(0.f, 0.f, 0.f, 0.f);
}

// ------------------------------------------------------------------ SpMM ----
// F=128: one wave per edge, 4 floats per lane (float4 gather + 4 L2 atomics).
__global__ __launch_bounds__(256)
void k_spmm128(const float* __restrict__ src, float* __restrict__ dst,
               const int* __restrict__ ei, const float* __restrict__ dis, int E) {
    long long tid = (long long)blockIdx.x * blockDim.x + threadIdx.x;
    int e = (int)(tid >> 5);
    if (e >= E) return;
    int lane = (int)(tid & 31);
    int d = ei[e];
    int s = ei[E + e];
    float w = dis[d] * dis[s];
    const float4 v = *(const float4*)(src + (size_t)s * 128 + lane * 4);
    float* o = dst + (size_t)d * 128 + lane * 4;
    atomicAdd(o + 0, v.x * w);
    atomicAdd(o + 1, v.y * w);
    atomicAdd(o + 2, v.z * w);
    atomicAdd(o + 3, v.w * w);
}

// F=16: 8 lanes per edge, float2 each.
__global__ __launch_bounds__(256)
void k_spmm16(const float* __restrict__ src, float* __restrict__ dst,
              const int* __restrict__ ei, const float* __restrict__ dis, int E) {
    long long tid = (long long)blockIdx.x * blockDim.x + threadIdx.x;
    int e = (int)(tid >> 3);
    if (e >= E) return;
    int sub = (int)(tid & 7);
    int d = ei[e];
    int s = ei[E + e];
    float w = dis[d] * dis[s];
    const float2 v = *(const float2*)(src + (size_t)s * 16 + sub * 2);
    float* o = dst + (size_t)d * 16 + sub * 2;
    atomicAdd(o + 0, v.x * w);
    atomicAdd(o + 1, v.y * w);
}

// ------------- fused self-loop + bias + ReLU + f32->f16 (next-layer input) --
__global__ void k_epi_relu_f16(const float* __restrict__ agg, const float* __restrict__ gem,
                               const float* __restrict__ dis, const float* __restrict__ bias,
                               _Float16* __restrict__ Hh, int total4) {
    int i4 = blockIdx.x * blockDim.x + threadIdx.x;
    if (i4 >= total4) return;
    size_t i = (size_t)i4 * 4;
    int node = (int)(i >> 7);          // F = 128
    int f    = (int)(i & 127);
    float w = dis[node]; w *= w;       // self-loop weight dis^2
    float4 a = *(const float4*)(agg + i);
    float4 g = *(const float4*)(gem + i);
    float4 b = *(const float4*)(bias + f);
    v4h h;
    h[0] = (_Float16)fmaxf(fmaf(g.x, w, a.x) + b.x, 0.f);
    h[1] = (_Float16)fmaxf(fmaf(g.y, w, a.y) + b.y, 0.f);
    h[2] = (_Float16)fmaxf(fmaf(g.z, w, a.z) + b.z, 0.f);
    h[3] = (_Float16)fmaxf(fmaf(g.w, w, a.w) + b.w, 0.f);
    *(v4h*)(Hh + i) = h;
}

// ------------------------- fused self-loop + bias + log_softmax (16 classes)
__global__ void k_final(const float* __restrict__ agg, const float* __restrict__ gem,
                        const float* __restrict__ dis, const float* __restrict__ bias,
                        float* __restrict__ out, int n) {
    int node = blockIdx.x * blockDim.x + threadIdx.x;
    if (node >= n) return;
    float w = dis[node]; w *= w;
    float z[16];
#pragma unroll
    for (int c = 0; c < 16; ++c)
        z[c] = fmaf(gem[(size_t)node * 16 + c], w, agg[(size_t)node * 16 + c]) + bias[c];
    float m = z[0];
#pragma unroll
    for (int c = 1; c < 16; ++c) m = fmaxf(m, z[c]);
    float s = 0.f;
#pragma unroll
    for (int c = 0; c < 16; ++c) s += __expf(z[c] - m);
    float l = m + __logf(s);
#pragma unroll
    for (int c = 0; c < 16; ++c) out[(size_t)node * 16 + c] = z[c] - l;
}

// ------------------------------------------------------------------ driver --
extern "C" void kernel_launch(void* const* d_in, const int* in_sizes, int n_in,
                              void* d_out, int out_size, void* d_ws, size_t ws_size,
                              hipStream_t stream) {
    const float* x  = (const float*)d_in[0];
    const int*   ei = (const int*)d_in[1];
    const float* W0 = (const float*)d_in[2];
    const float* b0 = (const float*)d_in[3];
    const float* W1 = (const float*)d_in[4];
    const float* b1 = (const float*)d_in[5];
    const float* W2 = (const float*)d_in[6];
    const float* b2 = (const float*)d_in[7];
    float* out = (float*)d_out;

    const int N = in_sizes[0] / NFEAT;
    const int E = in_sizes[1] / 2;

    // carve workspace (d_ws is 256B-aligned from hipMalloc)
    size_t off = 0;
    auto carve = [&](size_t bytes) -> char* {
        char* p = (char*)d_ws + off;
        off += (bytes + 255) & ~(size_t)255;
        return p;
    };
    float*    dis  = (float*)carve((size_t)N * 4);
    _Float16* Hh   = (_Float16*)carve((size_t)N * 128 * 2);
    _Float16* Wf   = (_Float16*)carve((size_t)128 * 128 * 2);
    float*    tmp  = (float*)carve((size_t)N * 128 * 4);
    float*    agg  = (float*)carve((size_t)N * 128 * 4);
    (void)ws_size;

    auto nb = [](long long total, int bs) { return (int)((total + bs - 1) / bs); };

    const int rowTiles  = (N + 15) / 16;
    const int groups128 = rowTiles * ((NHID / 16) / 4);   // CPW = 4 -> 2 groups/row
    const int groups16  = rowTiles;                       // CPW = 1, single col tile
    const long long NF   = (long long)N * 128;
    const long long NF16 = (long long)N * 16;

    // ---- degrees / normalization ----
    k_init_deg<<<nb(N, 256), 256, 0, stream>>>(dis, N);
    k_deg_accum<<<nb(E, 256), 256, 0, stream>>>(dis, ei, E);
    k_deg_rsqrt<<<nb(N, 256), 256, 0, stream>>>(dis, N);

    // ---- layer 0: H1 = relu(Ahat @ (x W0) + b0)  (H1 kept only as f16) ----
    k_f32_to_f16<<<nb(NF / 4, 256), 256, 0, stream>>>(x, Hh, (int)(NF / 4));
    k_pack_w<128><<<nb(8 * 4 * 32, 256), 256, 0, stream>>>(W0, Wf);
    k_gemm<128, 4><<<nb((long long)groups128 * 32, 256), 256, 0, stream>>>(Hh, Wf, tmp, N, groups128);
    k_zero4<<<nb(NF / 4, 256), 256, 0, stream>>>(agg, (int)(NF / 4));
    k_spmm128<<<nb((long long)E * 32, 256), 256, 0, stream>>>(tmp, agg, ei, dis, E);
    k_epi_relu_f16<<<nb(NF / 4, 256), 256, 0, stream>>>(agg, tmp, dis, b0, Hh, (int)(NF / 4));

    // ---- layer 1: H2 = relu(Ahat @ (H1 W1) + b1) ----
    k_pack_w<128><<<nb(8 * 4 * 32, 256), 256, 0, stream>>>(W1, Wf);
    k_gemm<128, 4><<<nb((long long)groups128 * 32, 256), 256, 0, stream>>>(Hh, Wf, tmp, N, groups128);
    k_zero4<<<nb(NF / 4, 256), 256, 0, stream>>>(agg, (int)(NF / 4));
    k_spmm128<<<nb((long long)E * 32, 256), 256, 0, stream>>>(tmp, agg, ei, dis, E);
    k_epi_relu_f16<<<nb(NF / 4, 256), 256, 0, stream>>>(agg, tmp, dis, b1, Hh, (int)(NF / 4));

    // ---- layer 2: out = log_softmax(Ahat @ (H2 W2) + b2) ----
    k_pack_w<16><<<nb(1 * 4 * 32, 128), 128, 0, stream>>>(W2, Wf);
    k_gemm<16, 1><<<nb((long long)groups16 * 32, 256), 256, 0, stream>>>(Hh, Wf, tmp, N, groups16);
    k_zero4<<<nb(NF16 / 4, 256), 256, 0, stream>>>(agg, (int)(NF16 / 4));
    k_spmm16<<<nb((long long)E * 8, 256), 256, 0, stream>>>(tmp, agg, ei, dis, E);
    k_final<<<nb(N, 256), 256, 0, stream>>>(agg, tmp, dis, b2, out, N);
}
